// NNDFMatcher_35218731827996
// MI455X (gfx1250) — matrix operand order repaired and checked by reading, not compiled, not run on hardware
//
#include <hip/hip_runtime.h>

typedef __attribute__((ext_vector_type(4)))  float  v4f;
typedef __attribute__((ext_vector_type(8)))  float  v8f;
typedef __attribute__((ext_vector_type(8)))  __bf16 v8bf;
typedef __attribute__((ext_vector_type(16))) __bf16 v16bf;

union BV16 {
  v16bf v;
  struct { v8bf lo; v8bf hi; } h;
};

#define BDIM 4
#define NDIM 4096
#define MDIM 4096
#define DDIM 128

// ---------------------------------------------------------------------------
// CDNA5 async copy global->LDS (ASYNCcnt-tracked), per CDNA5 ISA.
// LDS address = low 32 bits of the generic pointer (flat->LDS: addr[31:0]).
// ---------------------------------------------------------------------------
__device__ __forceinline__ void async_ld_b128(void* lds_dst, const void* gsrc) {
  unsigned lds_off = (unsigned)(unsigned long long)lds_dst;
  asm volatile("global_load_async_to_lds_b128 %0, %1, off"
               :: "v"(lds_off), "v"(gsrc) : "memory");
}
__device__ __forceinline__ void wait_asynccnt0() {
  asm volatile("s_wait_asynccnt 0x0" ::: "memory");
}

// ---------------------------------------------------------------------------
// Kernel 1: split fp32 rows into bf16 hi/lo pair + fp32 squared row norms.
// One 128-thread block per row (x rows first, then y rows). Inputs are
// read-once -> non-temporal loads.
// ---------------------------------------------------------------------------
__global__ void prep_split(const float* __restrict__ xd, const float* __restrict__ yd,
                           __bf16* __restrict__ xh, __bf16* __restrict__ xl,
                           __bf16* __restrict__ yh, __bf16* __restrict__ yl,
                           float* __restrict__ x2, float* __restrict__ y2) {
  int row = blockIdx.x;
  int t   = threadIdx.x;
  const float* src;
  __bf16 *hp, *lp;
  float* np;
  if (row < BDIM * NDIM) {
    src = xd + (size_t)row * DDIM;
    hp  = xh + (size_t)row * DDIM;
    lp  = xl + (size_t)row * DDIM;
    np  = x2 + row;
  } else {
    int r = row - BDIM * NDIM;
    src = yd + (size_t)r * DDIM;
    hp  = yh + (size_t)r * DDIM;
    lp  = yl + (size_t)r * DDIM;
    np  = y2 + r;
  }
  float v = __builtin_nontemporal_load(src + t);
  __bf16 hh = (__bf16)v;
  __bf16 ll = (__bf16)(v - (float)hh);
  hp[t] = hh;
  lp[t] = ll;
  float s = v * v;
#pragma unroll
  for (int off = 16; off; off >>= 1) s += __shfl_xor(s, off);
  __shared__ float sm[4];
  if ((t & 31) == 0) sm[t >> 5] = s;
  __syncthreads();
  if (t == 0) np[0] = sm[0] + sm[1] + sm[2] + sm[3];
}

// ---------------------------------------------------------------------------
// Kernel 2: zero the one-hot match_mask region (256 MB): write-once stream,
// non-temporal b128 stores so it does not evict the reused L2 working set.
// ---------------------------------------------------------------------------
__global__ void zero_out(v4f* __restrict__ p, long long n4) {
  long long i      = (long long)blockIdx.x * blockDim.x + threadIdx.x;
  long long stride = (long long)gridDim.x * blockDim.x;
  v4f z = {0.f, 0.f, 0.f, 0.f};
  for (; i < n4; i += stride) __builtin_nontemporal_store(z, p + i);
}

// ---------------------------------------------------------------------------
// Kernel 3: fused distance + argmin.
// Block = 8 waves = 128 n-rows; wave owns 16 rows (A tile register-resident,
// bf16 hi+lo). Per iteration: 32-row y tile staged to LDS via async b128
// copies (double-buffered), 2 sub-tiles x 4 K-chunks x 3 WMMAs = 24
// v_wmma_f32_16x16x32_bf16, then fp32 epilogue comparing on the squared key
// (sqrt deferred to the end), NT mask stream prefetched 4 tiles ahead.
// ---------------------------------------------------------------------------
__global__ __launch_bounds__(256) void nndf_matcher(
    const __bf16* __restrict__ xh, const __bf16* __restrict__ xl,
    const __bf16* __restrict__ yh, const __bf16* __restrict__ yl,
    const float* __restrict__ x2, const float* __restrict__ y2,
    const float* __restrict__ mask,
    int* __restrict__ pairs, float* __restrict__ topd) {
  __shared__ __align__(16) char ybuf[2][16384];   // [hi 8KB | lo 8KB] per buffer

  const int t    = threadIdx.x;
  const int lane = t & 31;
  const int li   = lane & 15;
  const int half = lane >> 4;
  const int wave = t >> 5;

  const int b     = blockIdx.x >> 5;              // 32 blocks per batch
  const int nblk  = blockIdx.x & 31;
  const int ntile = nblk * 128 + wave * 16;

  // --- A tile: 16 rows of x, hi+lo, ISA 16-bit A layout -------------------
  const size_t arow = ((size_t)b * NDIM + ntile + li) * DDIM;
  BV16 ah[4], al[4];
#pragma unroll
  for (int c = 0; c < 4; ++c) {
    const int d0 = 32 * c + 8 * half;
    ah[c].h.lo = *(const v8bf*)(xh + arow + d0);
    ah[c].h.hi = *(const v8bf*)(xh + arow + d0 + 16);
    al[c].h.lo = *(const v8bf*)(xl + arow + d0);
    al[c].h.hi = *(const v8bf*)(xl + arow + d0 + 16);
  }
  float x2v[8];
  const float* x2p = x2 + (size_t)b * NDIM + ntile + 8 * half;
#pragma unroll
  for (int v = 0; v < 8; ++v) x2v[v] = x2p[v];

  float best[8];                                   // squared key: d2 * m^2
  int   bidx[8];
#pragma unroll
  for (int v = 0; v < 8; ++v) { best[v] = 3.4e38f; bidx[v] = 0; }

  const float* maskb = mask + ((size_t)b * NDIM + ntile) * MDIM;
  const float* y2b   = y2 + (size_t)b * MDIM;
  const size_t ybase = (size_t)b * MDIM * DDIM;

  // stage(): 256 threads x 4 async b128 = 16KB (32 rows hi + 32 rows lo)
  auto stage = [&](char* buf, int mtile) {
#pragma unroll
    for (int j = 0; j < 2; ++j) {
      const int q   = t + j * 256;        // 16B chunk id in 8KB half
      const int row = q >> 4;             // 0..31
      const int col = (q & 15) * 8;       // bf16 element offset in row
      const size_t g = ybase + (size_t)(mtile + row) * DDIM + col;
      async_ld_b128(buf + q * 16,        yh + g);
      async_ld_b128(buf + 8192 + q * 16, yl + g);
    }
  };

  stage(ybuf[0], 0);
  wait_asynccnt0();
  __syncthreads();

  const int ITERS = MDIM / 32;                    // 128
  for (int i = 0; i < ITERS; ++i) {
    char* cur = ybuf[i & 1];
    if (i + 1 < ITERS) stage(ybuf[(i + 1) & 1], (i + 1) * 32);

    const int mtile = i * 32;
#pragma unroll
    for (int s = 0; s < 2; ++s) {
      // --- B fragments from LDS (ISA 16-bit B layout): row = s*16+li,
      //     lane<16 K 0..7,16..23 per chunk, lane>=16 K 8..15,24..31
      const size_t hb = (size_t)((s * 16 + li) * DDIM + 8 * half) * 2;
      v8f acc = {};
#pragma unroll
      for (int c = 0; c < 4; ++c) {
        const size_t co = hb + (size_t)(64 * c);  // 32 bf16 * 2B per chunk
        BV16 bh, bl;
        bh.h.lo = *(const v8bf*)(cur + co);
        bh.h.hi = *(const v8bf*)(cur + co + 32);
        bl.h.lo = *(const v8bf*)(cur + 8192 + co);
        bl.h.hi = *(const v8bf*)(cur + 8192 + co + 32);
        acc = __builtin_amdgcn_wmma_f32_16x16x32_bf16(false, ah[c].v, false, bh.v,
                                                      (short)0, acc, false, false);
        acc = __builtin_amdgcn_wmma_f32_16x16x32_bf16(false, ah[c].v, false, bl.v,
                                                      (short)0, acc, false, false);
        acc = __builtin_amdgcn_wmma_f32_16x16x32_bf16(false, al[c].v, false, bh.v,
                                                      (short)0, acc, false, false);
      }
      const int   m   = mtile + s * 16 + li;
      const float y2v = y2b[m];
#pragma unroll
      for (int v = 0; v < 8; ++v) {
        // coalesced NT mask stream; prefetch 4 tiles (~700ns) ahead into GL2
        const float mval = __builtin_nontemporal_load(
            &maskb[(size_t)(v + 8 * half) * MDIM + m]);
        __builtin_prefetch(&maskb[(size_t)(v + 8 * half) * MDIM + m + 128], 0, 0);
        float d2 = x2v[v] + y2v - 2.0f * acc[v];
        d2 = fmaxf(d2, 0.0f);
        float key = d2 * mval * mval;             // argmin-equivalent (m >= 0)
        if (key < best[v]) { best[v] = key; bidx[v] = m; }  // strict <: first min
      }
    }
    wait_asynccnt0();
    __syncthreads();
  }

  // --- min+argmin reduction across each 16-lane group ---------------------
#pragma unroll
  for (int v = 0; v < 8; ++v) {
    float bv = best[v];
    int   bi = bidx[v];
#pragma unroll
    for (int off = 8; off; off >>= 1) {
      float ov = __shfl_xor(bv, off);
      int   oi = __shfl_xor(bi, off);
      if (ov < bv || (ov == bv && oi < bi)) { bv = ov; bi = oi; }
    }
    if (li == 0) {  // lanes 0 and 16 -> rows ntile+v and ntile+v+8
      const int n = ntile + v + 8 * half;
      pairs[(size_t)b * NDIM + n] = bi;
      topd [(size_t)b * NDIM + n] = __builtin_sqrtf(bv);  // sqrt(d2*m^2)
    }
  }
}

// ---------------------------------------------------------------------------
// Kernel 4: scatter the one-hot 1.0f at argmin positions.
// ---------------------------------------------------------------------------
__global__ void scatter_ones(const int* __restrict__ pairs, float* __restrict__ mm) {
  int i = blockIdx.x * blockDim.x + threadIdx.x;
  if (i < BDIM * NDIM) mm[(size_t)i * MDIM + pairs[i]] = 1.0f;
}

extern "C" void kernel_launch(void* const* d_in, const int* in_sizes, int n_in,
                              void* d_out, int out_size, void* d_ws, size_t ws_size,
                              hipStream_t stream) {
  const float* xd   = (const float*)d_in[0];
  const float* yd   = (const float*)d_in[1];
  const float* mask = (const float*)d_in[2];

  float* out = (float*)d_out;
  const size_t BNM = (size_t)BDIM * NDIM * MDIM;   // 67,108,864
  const size_t BN  = (size_t)BDIM * NDIM;          // 16,384
  float* mm    = out;                               // match_mask
  int*   pairs = (int*)(out + BNM);                 // pairs (int32)
  float* topd  = out + BNM + BN;                    // top_dists

  // workspace layout: xh | xl | yh | yl (4 MB each) | x2 | y2
  char* ws = (char*)d_ws;
  const size_t SEG = (size_t)1 << 22;               // 4 MB = B*N*D bf16
  __bf16* xh = (__bf16*)(ws);
  __bf16* xl = (__bf16*)(ws + SEG);
  __bf16* yh = (__bf16*)(ws + 2 * SEG);
  __bf16* yl = (__bf16*)(ws + 3 * SEG);
  float*  x2 = (float*)(ws + 4 * SEG);
  float*  y2 = (float*)(ws + 4 * SEG + BN * sizeof(float));

  prep_split<<<BDIM * (NDIM + MDIM), 128, 0, stream>>>(xd, yd, xh, xl, yh, yl, x2, y2);
  zero_out<<<4096, 256, 0, stream>>>((v4f*)mm, (long long)(BNM / 4));
  nndf_matcher<<<BDIM * (NDIM / 128), 256, 0, stream>>>(xh, xl, yh, yl, x2, y2,
                                                        mask, pairs, topd);
  scatter_ones<<<(BDIM * NDIM + 255) / 256, 256, 0, stream>>>(pairs, mm);
}
